// ChemGCLayer_71545565216996
// MI455X (gfx1250) — matrix operand order repaired and checked by reading, not compiled, not run on hardware
//
#include <hip/hip_runtime.h>
#include <hip/hip_bf16.h>

// ---------------------------------------------------------------------------
// ChemGCLayer fused forward for MI455X (gfx1250, wave32, WMMA).
//
//   k0  pack_weights      : f32 weights -> bf16 B-fragment-swizzled layout (ws)
//   k1a deg_init          : deg[i] = 1 (self loop)
//   k1b deg_accum         : deg[dst] += 1 over edges (atomic)
//   k1c deg_rsqrt         : dinv = rsqrt(deg)
//   k2  node_forward(WMMA): nfeats = elu(elu(feats@W1+b1)@W2+b2)
//                           xs = (concat(nf,feats)@Wgc)*dinv      -> ws
//                           S  = xs (self-loop init)              -> ws
//                           partial = nf@Wc_top + bc              -> ws
//   k3  edge_scatter      : S[dst] += xs[src]   (1 wave / edge, f32 atomics)
//   k4  final_forward(WMMA): out = elu(partial + (dinv*S + bgc)@Wc_bot)
//   + async d2d copies of edges/batch into the output tuple tail.
//
// All C-layout <-> memory traffic goes through per-wave LDS slabs so global
// accesses are coalesced b128; ELU is branchless (v_exp_f32 co-executes with
// WMMA). edges assumed int32 (default jax config, x64 disabled).
// ---------------------------------------------------------------------------

typedef __attribute__((ext_vector_type(16))) __bf16 v16bf;
typedef __attribute__((ext_vector_type(8)))  float  v8f;

#define HID 128
#define INF 64

static __device__ __forceinline__ int imin(int a, int b) { return a < b ? a : b; }

// Branchless ELU (alpha=1): max(x,0) + exp(min(x,0)) - 1.
static __device__ __forceinline__ float elu1(float x) {
  return fmaxf(x, 0.f) + __expf(fminf(x, 0.f)) - 1.f;
}

static __device__ __forceinline__ v8f wmma_bf16(v16bf a, v16bf b, v8f c) {
  // D = A(16x32 bf16) * B(32x16 bf16) + C(16x16 f32)
  return __builtin_amdgcn_wmma_f32_16x16x32_bf16(false, a, false, b, (short)0, c,
                                                 false, false);
}

// A fragment (16x32 bf16) per ISA layout: lane m=lane&15 holds row m.
// lanes 0-15: elems 0..7 -> K 0..7,  8..15 -> K 16..23
// lanes16-31: elems 0..7 -> K 8..15, 8..15 -> K 24..31
static __device__ __forceinline__ v16bf a_frag_global(const float* base, int ld,
                                                      int row0, int k0, int n) {
  const int lane = threadIdx.x & 31;
  const int m = lane & 15, kh = lane >> 4;
  const float* p = base + (size_t)imin(row0 + m, n - 1) * ld + k0;
  v16bf a;
#pragma unroll
  for (int i = 0; i < 8; ++i) a[i] = (__bf16)p[8 * kh + i];
#pragma unroll
  for (int i = 0; i < 8; ++i) a[8 + i] = (__bf16)p[16 + 8 * kh + i];
  return a;
}

static __device__ __forceinline__ v16bf a_frag_lds(const __bf16* tile, int k0) {
  const int lane = threadIdx.x & 31;
  const int m = lane & 15, kh = lane >> 4;
  const __bf16* p = tile + m * HID + k0;
  v16bf a;
#pragma unroll
  for (int i = 0; i < 8; ++i) a[i] = p[8 * kh + i];
#pragma unroll
  for (int i = 0; i < 8; ++i) a[8 + i] = p[16 + 8 * kh + i];
  return a;
}

// B fragment: packed as frag=(kt*8+nt), 512 bf16 per fragment, 16 per lane
// (pre-swizzled by pack_weights -> one contiguous 32B load per lane).
static __device__ __forceinline__ v16bf b_frag(const __bf16* packed, int kt, int nt) {
  const int lane = threadIdx.x & 31;
  return *(const v16bf*)(packed + ((size_t)(kt * 8 + nt)) * 512 + lane * 16);
}

// ---- k0: swizzle f32 weight [K x 128] into bf16 B-fragment layout -----------
__global__ void pack_weights(const float* __restrict__ W, __bf16* __restrict__ dst,
                             int K) {
  const int t = blockIdx.x * blockDim.x + threadIdx.x;
  const int total = (K / 32) * 8 * 512;
  if (t >= total) return;
  const int frag = t >> 9;
  const int lane = (t >> 4) & 31;
  const int i = t & 15;
  const int kt = frag >> 3, nt = frag & 7;
  const int ncol = nt * 16 + (lane & 15);
  const int k = kt * 32 + (i & 7) + 8 * (lane >> 4) + 16 * (i >> 3);  // B: elem i -> K = (i&7) + 8*half + 16*(i>>3)
  dst[t] = (__bf16)W[(size_t)k * HID + ncol];
}

// ---- k1: degrees ------------------------------------------------------------
__global__ void deg_init(float* deg, int n) {
  int i = blockIdx.x * blockDim.x + threadIdx.x;
  if (i < n) deg[i] = 1.0f;  // self loop
}
__global__ void deg_accum(const int* __restrict__ edges, float* deg, int e) {
  int i = blockIdx.x * blockDim.x + threadIdx.x;
  if (i < e) atomicAdd(&deg[edges[e + i]], 1.0f);  // dst row
}
__global__ void deg_rsqrt(float* deg, int n) {
  int i = blockIdx.x * blockDim.x + threadIdx.x;
  if (i < n) deg[i] = rsqrtf(deg[i]);
}

// ---- k2: fused node-wise GEMM chain (WMMA) ---------------------------------
__global__ __launch_bounds__(128) void node_forward(
    const float* __restrict__ feats, const float* __restrict__ dinv,
    const __bf16* __restrict__ pW1, const float* __restrict__ b1,
    const __bf16* __restrict__ pW2, const float* __restrict__ b2,
    const __bf16* __restrict__ pWgc,
    const __bf16* __restrict__ pWcTop, const float* __restrict__ bc,
    float* __restrict__ xs, float* __restrict__ S, float* __restrict__ partial,
    int n) {
  __shared__ __bf16 stage[4][16 * HID];   // per-wave bf16 C->A relayout (16KB)
  __shared__ float  stageF[4][16 * HID];  // per-wave f32 store staging (32KB)
  const int wave = threadIdx.x >> 5;
  const int lane = threadIdx.x & 31;
  const int row0 = (blockIdx.x * 4 + wave) * 16;
  if (row0 >= n) return;  // no block barriers: each wave owns its LDS slabs

  const int m = lane & 15;
  const int half = lane >> 4;
  __bf16* st = stage[wave];
  float* stF = stageF[wave];

  v16bf aF0 = a_frag_global(feats, INF, row0, 0, n);
  v16bf aF1 = a_frag_global(feats, INF, row0, 32, n);

  // h1 = elu(feats@W1 + b1) -> LDS (bf16)
#pragma unroll
  for (int nt = 0; nt < 8; ++nt) {
    v8f acc = {};
    acc = wmma_bf16(aF0, b_frag(pW1, 0, nt), acc);
    acc = wmma_bf16(aF1, b_frag(pW1, 1, nt), acc);
    const int col = nt * 16 + m;
    const float bias = b1[col];
#pragma unroll
    for (int r = 0; r < 8; ++r)
      st[(r + 8 * half) * HID + col] = (__bf16)elu1(acc[r] + bias);
  }

  v16bf aH[4];
#pragma unroll
  for (int kt = 0; kt < 4; ++kt) aH[kt] = a_frag_lds(st, kt * 32);

  // nf = elu(h1@W2 + b2) -> LDS (bf16)
#pragma unroll
  for (int nt = 0; nt < 8; ++nt) {
    v8f acc = {};
#pragma unroll
    for (int kt = 0; kt < 4; ++kt) acc = wmma_bf16(aH[kt], b_frag(pW2, kt, nt), acc);
    const int col = nt * 16 + m;
    const float bias = b2[col];
#pragma unroll
    for (int r = 0; r < 8; ++r)
      st[(r + 8 * half) * HID + col] = (__bf16)elu1(acc[r] + bias);
  }

  v16bf aN[4];
#pragma unroll
  for (int kt = 0; kt < 4; ++kt) aN[kt] = a_frag_lds(st, kt * 32);

  float dvr[8];
#pragma unroll
  for (int r = 0; r < 8; ++r) dvr[r] = dinv[imin(row0 + r + 8 * half, n - 1)];

  // xs = (nf@Wgc[0:128] + feats@Wgc[128:192]) * dinv -> LDS f32
#pragma unroll
  for (int nt = 0; nt < 8; ++nt) {
    v8f acc = {};
#pragma unroll
    for (int kt = 0; kt < 4; ++kt) acc = wmma_bf16(aN[kt], b_frag(pWgc, kt, nt), acc);
    acc = wmma_bf16(aF0, b_frag(pWgc, 4, nt), acc);
    acc = wmma_bf16(aF1, b_frag(pWgc, 5, nt), acc);
    const int col = nt * 16 + m;
#pragma unroll
    for (int r = 0; r < 8; ++r)
      stF[(r + 8 * half) * HID + col] = acc[r] * dvr[r];
  }
  // Coalesced b128 stores: one 512B row per pass (xs and S self-loop init).
#pragma unroll 4
  for (int t = 0; t < 16; ++t) {
    const int gr = row0 + t;
    if (gr < n) {
      const float4 v = *(const float4*)(stF + t * HID + lane * 4);
      *(float4*)(xs + (size_t)gr * HID + lane * 4) = v;
      *(float4*)(S + (size_t)gr * HID + lane * 4) = v;
    }
  }

  // partial = nf@Wc_top + bc -> LDS f32 -> coalesced b128 stores
#pragma unroll
  for (int nt = 0; nt < 8; ++nt) {
    v8f acc = {};
#pragma unroll
    for (int kt = 0; kt < 4; ++kt)
      acc = wmma_bf16(aN[kt], b_frag(pWcTop, kt, nt), acc);
    const int col = nt * 16 + m;
    const float bias = bc[col];
#pragma unroll
    for (int r = 0; r < 8; ++r)
      stF[(r + 8 * half) * HID + col] = acc[r] + bias;
  }
#pragma unroll 4
  for (int t = 0; t < 16; ++t) {
    const int gr = row0 + t;
    if (gr < n)
      *(float4*)(partial + (size_t)gr * HID + lane * 4) =
          *(const float4*)(stF + t * HID + lane * 4);
  }
}

// ---- k3: edge gather/scatter (memory-bound; one wave per edge) -------------
__global__ void edge_scatter(const int* __restrict__ edges,
                             const float* __restrict__ xs, float* __restrict__ S,
                             int e) {
  const int gid = blockIdx.x * blockDim.x + threadIdx.x;
  const int eidx = gid >> 5;
  const int lane = gid & 31;
  if (eidx >= e) return;
  // Cover the (uniform) index-load latency of upcoming edges.
  __builtin_prefetch((const void*)(edges + eidx + 4096), 0, 0);
  __builtin_prefetch((const void*)(edges + e + eidx + 4096), 0, 0);
  const int src = edges[eidx];
  const int dst = edges[e + eidx];
  const float4 v = *(const float4*)(xs + (size_t)src * HID + lane * 4);
  float* d = S + (size_t)dst * HID + lane * 4;
  atomicAdd(d + 0, v.x);
  atomicAdd(d + 1, v.y);
  atomicAdd(d + 2, v.z);
  atomicAdd(d + 3, v.w);
}

// ---- k4: out = elu(partial + (dinv*S + bgc)@Wc_bot) (WMMA) -----------------
__global__ __launch_bounds__(128) void final_forward(
    const float* __restrict__ S, const float* __restrict__ dinv,
    const float* __restrict__ bgc, const __bf16* __restrict__ pWcBot,
    const float* __restrict__ partial, float* __restrict__ out, int n) {
  __shared__ float stageF[4][16 * HID];  // per-wave f32 staging (32KB)
  const int wave = threadIdx.x >> 5;
  const int lane = threadIdx.x & 31;
  const int row0 = (blockIdx.x * 4 + wave) * 16;
  if (row0 >= n) return;

  const int m = lane & 15;
  const int half = lane >> 4;
  float* stF = stageF[wave];

  // Coalesced b128 loads of partial -> LDS (C-layout reads come from LDS).
#pragma unroll 4
  for (int t = 0; t < 16; ++t) {
    const int gr = imin(row0 + t, n - 1);
    *(float4*)(stF + t * HID + lane * 4) =
        *(const float4*)(partial + (size_t)gr * HID + lane * 4);
  }

  // Build A fragments of gc = dinv*S + bgc directly (two b128 runs per kt).
  const int ar = imin(row0 + m, n - 1);
  const float dm = dinv[ar];
  v16bf aG[4];
#pragma unroll
  for (int kt = 0; kt < 4; ++kt) {
    const float* p = S + (size_t)ar * HID + kt * 32;
    v16bf a;
#pragma unroll
    for (int i = 0; i < 8; ++i) {
      const int k = 8 * half + i;
      a[i] = (__bf16)(dm * p[k] + bgc[kt * 32 + k]);
    }
#pragma unroll
    for (int i = 0; i < 8; ++i) {
      const int k = 16 + 8 * half + i;
      a[8 + i] = (__bf16)(dm * p[k] + bgc[kt * 32 + k]);
    }
    aG[kt] = a;
  }

#pragma unroll
  for (int nt = 0; nt < 8; ++nt) {
    const int col = nt * 16 + m;
    v8f acc;
#pragma unroll
    for (int r = 0; r < 8; ++r) acc[r] = stF[(r + 8 * half) * HID + col];
#pragma unroll
    for (int kt = 0; kt < 4; ++kt)
      acc = wmma_bf16(aG[kt], b_frag(pWcBot, kt, nt), acc);
    // Write results back to the same (disjoint per-nt) columns of the slab.
#pragma unroll
    for (int r = 0; r < 8; ++r) stF[(r + 8 * half) * HID + col] = acc[r];
  }

  // Coalesced b128 stores with branchless ELU applied on the way out.
#pragma unroll 4
  for (int t = 0; t < 16; ++t) {
    const int gr = row0 + t;
    if (gr < n) {
      float4 v = *(const float4*)(stF + t * HID + lane * 4);
      v.x = elu1(v.x);
      v.y = elu1(v.y);
      v.z = elu1(v.z);
      v.w = elu1(v.w);
      *(float4*)(out + (size_t)gr * HID + lane * 4) = v;
    }
  }
}

// ---------------------------------------------------------------------------
extern "C" void kernel_launch(void* const* d_in, const int* in_sizes, int n_in,
                              void* d_out, int out_size, void* d_ws, size_t ws_size,
                              hipStream_t stream) {
  const float* feats = (const float*)d_in[0];
  const int* edges = (const int*)d_in[1];  // [2,E] int32
  const int* batch = (const int*)d_in[2];
  const float* W1 = (const float*)d_in[3];
  const float* b1 = (const float*)d_in[4];
  const float* W2 = (const float*)d_in[5];
  const float* b2 = (const float*)d_in[6];
  const float* Wgc = (const float*)d_in[7];
  const float* bgc = (const float*)d_in[8];
  const float* Wc = (const float*)d_in[9];
  const float* bc = (const float*)d_in[10];

  const int n = in_sizes[0] / INF;  // 100000
  const int e = in_sizes[1] / 2;    // 1600000

  // Workspace carve-up (~154 MB): packed bf16 weights + deg/dinv + xs/S/partial.
  char* ws = (char*)d_ws;
  size_t o = 0;
  __bf16* pW1 = (__bf16*)(ws + o);   o += (size_t)16 * 1024;   // K=64
  __bf16* pW2 = (__bf16*)(ws + o);   o += (size_t)32 * 1024;   // K=128
  __bf16* pWgc = (__bf16*)(ws + o);  o += (size_t)48 * 1024;   // K=192
  __bf16* pWcT = (__bf16*)(ws + o);  o += (size_t)32 * 1024;   // Wc rows 0..127
  __bf16* pWcB = (__bf16*)(ws + o);  o += (size_t)32 * 1024;   // Wc rows 128..255
  float* deg = (float*)(ws + o);     o += (((size_t)n * 4) + 255) & ~(size_t)255;
  float* xs = (float*)(ws + o);      o += (size_t)n * HID * 4;
  float* S = (float*)(ws + o);       o += (size_t)n * HID * 4;
  float* partial = (float*)(ws + o); o += (size_t)n * HID * 4;
  (void)ws_size; (void)n_in;

  pack_weights<<<(16 * 512 + 255) / 256, 256, 0, stream>>>(W1, pW1, 64);
  pack_weights<<<(32 * 512 + 255) / 256, 256, 0, stream>>>(W2, pW2, 128);
  pack_weights<<<(48 * 512 + 255) / 256, 256, 0, stream>>>(Wgc, pWgc, 192);
  pack_weights<<<(32 * 512 + 255) / 256, 256, 0, stream>>>(Wc, pWcT, 128);
  pack_weights<<<(32 * 512 + 255) / 256, 256, 0, stream>>>(Wc + (size_t)128 * HID,
                                                           pWcB, 128);

  deg_init<<<(n + 255) / 256, 256, 0, stream>>>(deg, n);
  deg_accum<<<(e + 255) / 256, 256, 0, stream>>>(edges, deg, e);
  deg_rsqrt<<<(n + 255) / 256, 256, 0, stream>>>(deg, n);

  const int tiles = (n + 15) / 16;
  node_forward<<<(tiles + 3) / 4, 128, 0, stream>>>(feats, deg, pW1, b1, pW2, b2,
                                                    pWgc, pWcT, bc, xs, S, partial,
                                                    n);

  const long long ethreads = (long long)e * 32;
  edge_scatter<<<(int)((ethreads + 255) / 256), 256, 0, stream>>>(edges, xs, S, e);

  final_forward<<<(tiles + 3) / 4, 128, 0, stream>>>(S, deg, bgc, pWcB, partial,
                                                     (float*)d_out, n);

  // Output tuple tail: edges then batch (raw bit copies).
  char* outc = (char*)d_out;
  const size_t feats_bytes = (size_t)n * HID * sizeof(float);
  hipMemcpyAsync(outc + feats_bytes, (const void*)edges,
                 (size_t)e * 2 * sizeof(int), hipMemcpyDeviceToDevice, stream);
  hipMemcpyAsync(outc + feats_bytes + (size_t)e * 2 * sizeof(int),
                 (const void*)batch, (size_t)n * sizeof(int),
                 hipMemcpyDeviceToDevice, stream);
}
